// IOHMM_RNN_64476049047744
// MI455X (gfx1250) — matrix operand order repaired and checked by reading, not compiled, not run on hardware
//
#include <hip/hip_runtime.h>

// ---------------------------------------------------------------------------
// IOHMM RNN for gfx1250 (MI455X): f16 WMMA (v_wmma_f32_16x16x32_f16),
// 1 kernel per recurrent step (z_t GEMM fused with y_{t-1} emit GEMM),
// weights pre-packed into WMMA B-fragment lane layout, A staged via
// double-buffered LDS (one barrier per K-tile).
// B=256, T=512, NI=256, NS=512, NO=256, K_trans = NI+NS = 768.
// ---------------------------------------------------------------------------

typedef __attribute__((ext_vector_type(16))) _Float16     v16h;
typedef __attribute__((ext_vector_type(8)))  float        v8f;
typedef __attribute__((ext_vector_type(4)))  float        f32x4;
typedef __attribute__((ext_vector_type(4)))  unsigned int u32x4;

union FragU {
  v16h h;
  unsigned int u[8];
  u32x4 q[2];
};

__device__ __forceinline__ unsigned short f2h_bits(float a) {
  _Float16 h = (_Float16)a;
  unsigned short s;
  __builtin_memcpy(&s, &h, 2);
  return s;
}

// One v_cvt_pk_rtz_f16_f32 per dword (no scalar re-assembly noise).
__device__ __forceinline__ unsigned int pack2h(float a, float b) {
  auto h = __builtin_amdgcn_cvt_pkrtz(a, b);
  unsigned int u;
  __builtin_memcpy(&u, &h, 4);
  return u;
}

// ---------------------------------------------------------------------------
// Pack a row-major fp32 weight matrix W (N rows, K cols, leading dim ld) into
// f16 WMMA B-operand fragments. B[k][n] = W[n][k].
// B 32x16 layout used here (contiguous k): lane lo = N, hi = lane>>4 selects K half:
//   dword v holds K = {2v + 16*hi, 2v+1 + 16*hi}  (low half first).
// Storage: out[tile*256 + lane*8 + v], tile = nt*ktiles + kt  (K-major walk).
// One wave per tile, 8 tiles per 256-thread block.
// ---------------------------------------------------------------------------
__global__ __launch_bounds__(256) void k_pack(const float* __restrict__ W,
                                              unsigned int* __restrict__ out,
                                              int ktiles, int ld) {
  const int tid  = threadIdx.x;
  const int wave = tid >> 5, lane = tid & 31, lo = lane & 15, hi = lane >> 4;
  const int tileIdx = blockIdx.x * 8 + wave;
  const int nt = tileIdx / ktiles, kt = tileIdx % ktiles;
  const float* src = W + (size_t)(nt * 16 + lo) * ld + kt * 32 + 16 * hi;
  unsigned int* dst = out + (size_t)tileIdx * 256 + lane * 8;
#pragma unroll
  for (int v = 0; v < 8; ++v) {
    float a = src[2 * v];
    float b = src[2 * v + 1];
    dst[v] = pack2h(a, b);
  }
}

// ---------------------------------------------------------------------------
// Stage one 64x32 f16 A-block into an LDS buffer (20-dword row pitch:
// 16B-aligned ds_store_b128, conflict-free fragment reads).
//   kk <  xkt : rows from fp32 source (x), cvt_pk_rtz to f16, non-temporal
//   kk >= xkt : rows from f16 source (z buffer), straight 16B copy
// ---------------------------------------------------------------------------
__device__ __forceinline__ void stage_a(unsigned int* buf, int kk,
                                        const float* xsrc, int xstride,
                                        const unsigned short* zsrc, int zstride,
                                        int xkt, int arow, int aseg) {
  if (kk < xkt) {
    const f32x4* p =
        reinterpret_cast<const f32x4*>(xsrc + (size_t)arow * xstride + kk * 32 + aseg * 8);
    f32x4 f0 = __builtin_nontemporal_load(p);
    f32x4 f1 = __builtin_nontemporal_load(p + 1);
    u32x4 d;
    d[0] = pack2h(f0[0], f0[1]);
    d[1] = pack2h(f0[2], f0[3]);
    d[2] = pack2h(f1[0], f1[1]);
    d[3] = pack2h(f1[2], f1[3]);
    *reinterpret_cast<u32x4*>(buf + arow * 20 + aseg * 4) = d;
  } else {
    const u32x4* p = reinterpret_cast<const u32x4*>(
        zsrc + (size_t)arow * zstride + (kk - xkt) * 32 + aseg * 8);
    *reinterpret_cast<u32x4*>(buf + arow * 20 + aseg * 4) = *p;
  }
}

// ---------------------------------------------------------------------------
// Core 64x64 block GEMM, K walked in 32-wide tiles, double-buffered LDS.
// Each wave: 1 A-frag (16 rows; contiguous k order reads dwords
// {0..7}+8*hi -> two aligned ds_load_b128), 2 B-frags
// (2x global_load_b128 each from the pre-packed buffer), 2 wmma per K-tile,
// one workgroup barrier per K-tile.
// ---------------------------------------------------------------------------
__device__ __forceinline__ void gemm_core(unsigned int* lds,  // 2 * 1280 dwords
                                          const float* xsrc, int xstride,
                                          const unsigned short* zsrc, int zstride,
                                          int xkt, int ktiles,
                                          const unsigned int* __restrict__ Bp,
                                          int nt0, v8f& c0, v8f& c1) {
  const int tid  = threadIdx.x;
  const int lane = tid & 31, lo = lane & 15, hi = lane >> 4;
  const int waveM = (tid >> 5) >> 1;
  const int arow = tid >> 2, aseg = tid & 3;

  stage_a(lds, 0, xsrc, xstride, zsrc, zstride, xkt, arow, aseg);
  __syncthreads();

  for (int kk = 0; kk < ktiles; ++kk) {
    if (kk + 1 < ktiles)
      stage_a(lds + ((kk + 1) & 1) * 1280, kk + 1, xsrc, xstride, zsrc, zstride, xkt,
              arow, aseg);

    const unsigned int* cur = lds + (kk & 1) * 1280;
    FragU a;
    {
      const unsigned int* ab = cur + (waveM * 16 + lo) * 20 + 8 * hi;
      a.u[0] = ab[0];  a.u[1] = ab[1];  a.u[2] = ab[2];  a.u[3] = ab[3];
      a.u[4] = ab[4];  a.u[5] = ab[5];  a.u[6] = ab[6]; a.u[7] = ab[7];
    }
    FragU b;
    {
      const u32x4* bp =
          reinterpret_cast<const u32x4*>(Bp + ((size_t)nt0 * ktiles + kk) * 256 + lane * 8);
      b.q[0] = bp[0];
      b.q[1] = bp[1];
    }
    c0 = __builtin_amdgcn_wmma_f32_16x16x32_f16(false, a.h, false, b.h, (short)0, c0,
                                                false, false);
    {
      const u32x4* bp = reinterpret_cast<const u32x4*>(
          Bp + ((size_t)(nt0 + 1) * ktiles + kk) * 256 + lane * 8);
      b.q[0] = bp[0];
      b.q[1] = bp[1];
    }
    c1 = __builtin_amdgcn_wmma_f32_16x16x32_f16(false, a.h, false, b.h, (short)0, c1,
                                                false, false);
    __syncthreads();
  }
}

// Epilogue helpers. C/D layout: VGPR r -> row M = r + 8*hi, col N = lo.
__device__ __forceinline__ void store_z_sigmoid(const v8f& c0, const v8f& c1,
                                                const float* __restrict__ bias,
                                                unsigned short* __restrict__ z,
                                                int blockM0, int blockN0, int waveM,
                                                int waveN, int lo, int hi) {
#pragma unroll
  for (int nsub = 0; nsub < 2; ++nsub) {
    const v8f& c = nsub ? c1 : c0;
    const int col = blockN0 + waveN * 32 + nsub * 16 + lo;
    const float bv = bias[col];
#pragma unroll
    for (int r = 0; r < 8; ++r) {
      const int row = blockM0 + waveM * 16 + r + 8 * hi;
      float v = c[r] + bv;
      v = 1.0f / (1.0f + __expf(-v));
      z[row * 512 + col] = f2h_bits(v);
    }
  }
}

__device__ __forceinline__ void store_y(const v8f& c0, const v8f& c1,
                                        const float* __restrict__ bias,
                                        float* __restrict__ y, int t, int blockM0,
                                        int blockN0, int waveM, int waveN, int lo,
                                        int hi) {
#pragma unroll
  for (int nsub = 0; nsub < 2; ++nsub) {
    const v8f& c = nsub ? c1 : c0;
    const int col = blockN0 + waveN * 32 + nsub * 16 + lo;
    const float bv = bias[col];
#pragma unroll
    for (int r = 0; r < 8; ++r) {
      const int row = blockM0 + waveM * 16 + r + 8 * hi;  // batch
      __builtin_nontemporal_store(c[r] + bv,
                                  &y[(size_t)row * 131072 + (size_t)t * 256 + col]);
    }
  }
}

// z0 = sigmoid(x[:,0] @ W_init.T + b_init). Grid 32 = 4(M) x 8(N). K=256.
__global__ __launch_bounds__(256) void k_init(const float* __restrict__ x,
                                              const unsigned int* __restrict__ Wp,
                                              const float* __restrict__ bias,
                                              unsigned short* __restrict__ z0) {
  __shared__ __align__(16) unsigned int lds[2 * 64 * 20];
  const int tid = threadIdx.x, lane = tid & 31, lo = lane & 15, hi = lane >> 4;
  const int wave = tid >> 5, waveM = wave >> 1, waveN = wave & 1;
  const int bm = blockIdx.x >> 3, bn = blockIdx.x & 7;
  const int blockM0 = bm * 64, blockN0 = bn * 64;
  v8f c0 = {}, c1 = {};
  gemm_core(lds, x + (size_t)blockM0 * 131072, 131072, nullptr, 0, 16, 16, Wp,
            bn * 4 + waveN * 2, c0, c1);
  store_z_sigmoid(c0, c1, bias, z0, blockM0, blockN0, waveM, waveN, lo, hi);
}

// Per-step kernel: blocks 0..31 compute z_t (K=768: x_t | z_{t-1});
// blocks 32..47 compute y_{t-1} = z_{t-1} @ W_emit.T (K=512).
__global__ __launch_bounds__(256) void k_step(
    const float* __restrict__ x, int t, const unsigned short* __restrict__ zprev,
    unsigned short* __restrict__ zcur, const unsigned int* __restrict__ Wtp,
    const unsigned int* __restrict__ Wep, const float* __restrict__ btrans,
    const float* __restrict__ bemit, float* __restrict__ y) {
  __shared__ __align__(16) unsigned int lds[2 * 64 * 20];
  const int tid = threadIdx.x, lane = tid & 31, lo = lane & 15, hi = lane >> 4;
  const int wave = tid >> 5, waveM = wave >> 1, waveN = wave & 1;
  const int bid = blockIdx.x;
  v8f c0 = {}, c1 = {};
  if (bid < 32) {
    const int bm = bid >> 3, bn = bid & 7;
    const int blockM0 = bm * 64, blockN0 = bn * 64;
    gemm_core(lds, x + (size_t)blockM0 * 131072 + t * 256, 131072,
              zprev + blockM0 * 512, 512, 8, 24, Wtp, bn * 4 + waveN * 2, c0, c1);
    store_z_sigmoid(c0, c1, btrans, zcur, blockM0, blockN0, waveM, waveN, lo, hi);
  } else {
    const int eb = bid - 32;
    const int bm = eb >> 2, bn = eb & 3;
    const int blockM0 = bm * 64, blockN0 = bn * 64;
    gemm_core(lds, nullptr, 0, zprev + blockM0 * 512, 512, 0, 16, Wep,
              bn * 4 + waveN * 2, c0, c1);
    store_y(c0, c1, bemit, y, t - 1, blockM0, blockN0, waveM, waveN, lo, hi);
  }
}

// Final emit: y_{T-1} = z_{T-1} @ W_emit.T. Grid 16 = 4(M) x 4(N).
__global__ __launch_bounds__(256) void k_emit(const unsigned short* __restrict__ z,
                                              const unsigned int* __restrict__ Wep,
                                              const float* __restrict__ bemit,
                                              float* __restrict__ y, int t) {
  __shared__ __align__(16) unsigned int lds[2 * 64 * 20];
  const int tid = threadIdx.x, lane = tid & 31, lo = lane & 15, hi = lane >> 4;
  const int wave = tid >> 5, waveM = wave >> 1, waveN = wave & 1;
  const int bm = blockIdx.x >> 2, bn = blockIdx.x & 3;
  const int blockM0 = bm * 64, blockN0 = bn * 64;
  v8f c0 = {}, c1 = {};
  gemm_core(lds, nullptr, 0, z + blockM0 * 512, 512, 0, 16, Wep, bn * 4 + waveN * 2,
            c0, c1);
  store_y(c0, c1, bemit, y, t, blockM0, blockN0, waveM, waveN, lo, hi);
}

// ---------------------------------------------------------------------------
// Workspace layout (uint32 units):
//   [0,        131072)  packed W_init   (16 ktiles x 32 ntiles)
//   [131072,   524288)  packed W_trans  (48 ktiles x 32 ntiles)
//   [524288,   655360)  packed W_emit   (32 ktiles x 16 ntiles)
//   [655360,   786432)  z ping-pong: 2 x (256*512) f16
// Total = 3 MB.
// ---------------------------------------------------------------------------
extern "C" void kernel_launch(void* const* d_in, const int* in_sizes, int n_in,
                              void* d_out, int out_size, void* d_ws, size_t ws_size,
                              hipStream_t stream) {
  const float* x      = (const float*)d_in[0];
  const float* Winit  = (const float*)d_in[1];
  const float* binit  = (const float*)d_in[2];
  const float* Wtrans = (const float*)d_in[3];
  const float* btrans = (const float*)d_in[4];
  const float* Wemit  = (const float*)d_in[5];
  const float* bemit  = (const float*)d_in[6];
  float* y = (float*)d_out;

  unsigned int* ws       = (unsigned int*)d_ws;
  unsigned int* Winit_p  = ws;
  unsigned int* Wtrans_p = ws + 131072;
  unsigned int* Wemit_p  = ws + 524288;
  unsigned short* zb     = (unsigned short*)(ws + 655360);  // 2 x 131072 halves

  // Pack weights into WMMA B-fragment layout (one-time per call).
  k_pack<<<64, 256, 0, stream>>>(Winit, Winit_p, 16, 256);
  k_pack<<<192, 256, 0, stream>>>(Wtrans, Wtrans_p, 48, 768);
  k_pack<<<64, 256, 0, stream>>>(Wemit, Wemit_p, 32, 512);

  // z0
  k_init<<<32, 256, 0, stream>>>(x, Winit_p, binit, zb);

  // Recurrence: step t computes z_t and emits y_{t-1}; stream order is the
  // inter-step barrier (graph replay keeps the chain).
  for (int t = 1; t < 512; ++t) {
    const unsigned short* zp = zb + ((t - 1) & 1) * 131072;
    unsigned short* zc = zb + (t & 1) * 131072;
    k_step<<<48, 256, 0, stream>>>(x, t, zp, zc, Wtrans_p, Wemit_p, btrans, bemit, y);
  }

  // y_{511}
  k_emit<<<16, 256, 0, stream>>>(zb + 131072, Wemit_p, bemit, y, 511);
}
